// Model_82600811036996
// MI455X (gfx1250) — matrix operand order repaired and checked by reading, not compiled, not run on hardware
//
#include <hip/hip_runtime.h>
#include <hip/hip_bf16.h>

// ============================================================================
// MI455X (gfx1250) implementation.
// fp16 activations/weights via v_wmma_f32_16x16x32_f16, fp32 accumulate.
// All B-operands staged [K][N]; B-fragment loads are strided 16-bit loads (one per k).
// Layer-2 LSTM uses TDM (tensor_load_to_lds) double-buffered tile prefetch.
//
// d_in layout (setup_inputs dict order; params flattened jax-pytree style,
// dict keys sorted, lists in order):
//  0:y[512,128,1] 1:index 2:pos[512,128,10] 3:neg
//  attn:  4:lin.W[128,128] 5:lin.b
//         tr[0]: 6:W1[256,128] 7:W2[128,256] 8:Wi[384,128] 9:Wo[128,128]
//                10:b1 11:b2 12:be0 13:be1 14:bi 15:bo 16:g0 17:g1
//         18:v[128]
//  enc_tr: 19:W1 20:W2 21:Wi 22:Wo 23:b1 24:b2 25:be0 26:be1 27:bi 28:bo 29:g0 30:g1
//  lstm[0]: bwd 31:Whh[256,64] 32:Wih[256,1] 33:bhh 34:bih ; fwd 35..38
//  lstm[1]: bwd 39:Whh 40:Wih[256,128] 41:bhh 42:bih ; fwd 43..46
//  mi: 47:WCI 48:WCN 49:WCP 50:WCY 51:WF 52:WII 53:WIN 54:WIP 55:WIY 56:WO (each [128,256])
//      57:bAI 58:bAN 59:bAP 60:bAY
//      61:bCI 62:bCN 63:bCP 64:bCY 65:bF 66:bII 67:bIN 68:bIP 69:bIY 70:bO
//      71:wA[128,128]
//  reg: blocks[0]: 72:bn.b 73:bn.g 74:bn.rm 75:bn.rv 76:lin.W[512,128] 77:lin.b
//       blocks[1]: 78..81 bn, 82:lin.W[256,512] 83:lin.b
//       blocks[2]: 84..87 bn, 88:lin.W[128,256] 89:lin.b
//       bn0: 90:b 91:g 92:rm 93:rv ; out: 94:W[1,128] 95:b[1]
// ============================================================================

typedef _Float16 half_t;
typedef __attribute__((ext_vector_type(2)))  _Float16 h2;
typedef __attribute__((ext_vector_type(16))) _Float16 v16h;
typedef __attribute__((ext_vector_type(8)))  float    v8f;
typedef __attribute__((ext_vector_type(4)))  unsigned int u32x4;
typedef __attribute__((ext_vector_type(8)))  int i32x8;
typedef __attribute__((ext_vector_type(4)))  int i32x4;

#define BATCH 512
#define SEQ   128
#define NSTREAM 4
#define SB    (NSTREAM * BATCH)          // 2048
#define R1ROWS ((size_t)SB * SEQ)        // 262144
#define R2ROWS ((size_t)BATCH * SEQ)     // 65536

#define ACT_NONE 0
#define ACT_MISH 1
#define ACT_TANH 2

// ---------------------------------------------------------------------------
// WMMA helpers
// ---------------------------------------------------------------------------
__device__ inline v8f wmma16(v16h a, v16h b, v8f c) {
  return __builtin_amdgcn_wmma_f32_16x16x32_f16(false, a, false, b, (short)0, c,
                                                false, false);
}

// A-fragment (16x32) from row-major [m][k], stride ld (even). Contiguous-pair
// loads -> b32 accesses. Lanes 0-15: rows M, K 0..7 & 16..23; lanes 16-31: K+8.
__device__ inline v16h load_a_frag(const half_t* p, int ld, int m0, int k0) {
  int lane = threadIdx.x & 31;
  const h2* r2 = (const h2*)(p + (size_t)(m0 + (lane & 15)) * ld + k0 + ((lane >> 4) << 3));
  v16h a;
#pragma unroll
  for (int v = 0; v < 4; ++v) { h2 t = r2[v]; a[2 * v] = t.x; a[2 * v + 1] = t.y; }
#pragma unroll
  for (int v = 0; v < 4; ++v) { h2 t = r2[8 + v]; a[8 + 2 * v] = t.x; a[9 + 2 * v] = t.y; }
  return a;
}

// B-fragment (32x16) from [K][N] row-major storage (stride ld, even):
// lane -> N column; elements -> K 0..7 & 16..23 (lanes 16-31: K+8).
__device__ inline v16h load_b_frag_kn(const half_t* p, int ld, int k0, int n0) {
  int lane = threadIdx.x & 31;
  const half_t* col = p + (size_t)(k0 + ((lane >> 4) << 3)) * ld + n0 + (lane & 15);
  v16h b;
#pragma unroll
  for (int v = 0; v < 8; ++v) { b[v] = col[(size_t)v * ld]; b[8 + v] = col[(size_t)(16 + v) * ld]; }
  return b;
}

__device__ inline float sigf(float x) { return 1.f / (1.f + __expf(-x)); }
__device__ inline float actf(float x, int act) {
  if (act == ACT_MISH) { float sp = (x > 20.f) ? x : log1pf(__expf(x)); return x * tanhf(sp); }
  if (act == ACT_TANH) return tanhf(x);
  return x;
}

// ---------------------------------------------------------------------------
// TDM: async 2D tile load (16 rows x 128 halves) global->LDS, row stride in
// halves. D# fields per CDNA5 ISA: group0 = {flags, lds_addr, gaddr, type},
// group1 = {data_size, tensor dims, tile dims, stride}. 6-arg builtin
// (amdgpu-toolchain clang-23 signature).
// ---------------------------------------------------------------------------
__device__ inline void tdm_load_16x128(const half_t* gsrc, unsigned lds_byte,
                                       unsigned rowStrideHalves) {
  unsigned long long ga = (unsigned long long)(size_t)gsrc;
  u32x4 g0;
  g0[0] = 1u;                                              // count=1 (user D#)
  g0[1] = lds_byte;                                        // lds_addr (bytes)
  g0[2] = (unsigned)(ga & 0xFFFFFFFFu);                    // global_addr lo
  g0[3] = (unsigned)((ga >> 32) & 0x01FFFFFFu) | (2u << 30); // addr hi | type=2
  i32x8 g1;
  g1[0] = (1 << 16);                  // data_size = 1 -> 2 bytes
  g1[1] = (int)(128u << 16);          // tensor_dim0[15:0] in bits[63:48]
  g1[2] = (int)(16u << 16);           // tensor_dim1[15:0] in bits[95:80]
  g1[3] = (int)(128u << 16);          // tile_dim0 in bits[127:112]
  g1[4] = 16;                         // tile_dim1 in bits[143:128]
  g1[5] = (int)rowStrideHalves;       // tensor_dim0_stride lo32
  g1[6] = 0;
  g1[7] = 0;
  i32x4 z4 = {0, 0, 0, 0};
  i32x8 z8 = {0, 0, 0, 0, 0, 0, 0, 0};
  __builtin_amdgcn_tensor_load_to_lds(g0, g1, z4, z4, z8, 0);
}

// ---------------------------------------------------------------------------
// Generic GEMM:  C[:, coff:coff+N] = act(A[M,K]f16 * W^T + bias)
// W fp32 [N][ldw], row j at W + j*ldw + woff. Block 256 thr, tile 32x64,
// W tile staged transposed [K32][N64] in LDS.
// ---------------------------------------------------------------------------
__global__ void k_gemm(const half_t* __restrict__ A, int K,
                       const float* __restrict__ W, int ldw, int woff,
                       const float* __restrict__ bias,
                       half_t* __restrict__ C, int ldc, int coff,
                       int M, int N, int act) {
  __shared__ half_t As[32 * 36];
  __shared__ half_t Ws[32 * 68];     // [k][n] transposed
  int tid = threadIdx.x;
  int m0 = blockIdx.x * 32;
  int nB = blockIdx.y * 64;
  int w = tid >> 5, lane = tid & 31, hi = lane >> 4, lo = lane & 15;
  int mt = (w & 1) * 16, nt = (w >> 1) * 16;

  v8f acc = {};
  int nk = K >> 5;
  for (int kc = 0; kc < nk; ++kc) {
    int k0 = kc << 5;
    // stage A tile 32x32 (one 8-byte copy / thread)
    {
      int r = tid >> 3, ks = (tid & 7) << 2;
      const uint2* src = (const uint2*)(A + (size_t)(m0 + r) * K + k0 + ks);
      *(uint2*)&As[r * 36 + ks] = *src;
      if (kc + 1 < nk) __builtin_prefetch((const void*)(src + 8), 0, 0);
    }
    // stage W tile transposed: Ws[k][n], fp32->fp16 (8 floats / thread)
    {
      int n = tid >> 2, ks = (tid & 3) << 3;
      bool ok = (nB + n) < N;
      const float* src = W + (size_t)(nB + n) * ldw + woff + k0 + ks;
#pragma unroll
      for (int j = 0; j < 8; ++j) Ws[(ks + j) * 68 + n] = ok ? (half_t)src[j] : (half_t)0.f;
    }
    __syncthreads();
    v16h a = load_a_frag(As, 36, mt, 0);
    v16h b = load_b_frag_kn(Ws, 68, 0, nt);
    acc = wmma16(a, b, acc);
    __syncthreads();
  }
  int col = nB + nt + lo;
  if (col < N) {
    float bv = bias[col];
#pragma unroll
    for (int r = 0; r < 8; ++r) {
      int row = m0 + mt + r + hi * 8;
      C[(size_t)row * ldc + coff + col] = (half_t)actf(acc[r] + bv, act);
    }
  }
}

// ---------------------------------------------------------------------------
// Fused MHA for one (batch, head): Q*K^T -> softmax -> P*V, all in LDS.
// K is staged transposed [d][t] so the score B-fragments are contiguous.
// ---------------------------------------------------------------------------
__global__ void k_attn(const half_t* __restrict__ QKV, half_t* __restrict__ O, int T) {
  extern __shared__ char smem[];
  half_t* Qs  = (half_t*)smem;        // [128][40]
  half_t* Vs  = Qs + 128 * 40;        // [128][40]  ([t][d] = [K][N])
  half_t* Kst = Vs + 128 * 40;        // [32][136]  ([d][t] = [K][N])
  half_t* Ps  = Kst + 32 * 136;       // [128][136]
  int bh = blockIdx.x, b = bh >> 2, h = bh & 3;
  int tid = threadIdx.x;
  for (int e = tid; e < T * 32; e += 256) {
    int r = e >> 5, c = e & 31;
    size_t base = ((size_t)b * T + r) * 384 + h * 32 + c;
    Qs[r * 40 + c]  = QKV[base];
    Kst[c * 136 + r] = QKV[base + 128];
    Vs[r * 40 + c]  = QKV[base + 256];
  }
  __syncthreads();
  int w = tid >> 5, lane = tid & 31, hi = lane >> 4, lo = lane & 15;
  const float scale = 0.1767766952966369f;   // 1/sqrt(32)
  v16h qa = load_a_frag(Qs, 40, w * 16, 0);
  for (int kt = 0; kt < 8; ++kt) {
    v8f acc = {};
    acc = wmma16(qa, load_b_frag_kn(Kst, 136, 0, kt * 16), acc);
#pragma unroll
    for (int r = 0; r < 8; ++r)
      Ps[(w * 16 + r + hi * 8) * 136 + kt * 16 + lo] = (half_t)(acc[r] * scale);
  }
  __syncthreads();
  if (tid < T) {
    float mx = -1e30f;
    for (int j = 0; j < T; ++j) mx = fmaxf(mx, (float)Ps[tid * 136 + j]);
    float sum = 0.f;
    for (int j = 0; j < T; ++j) {
      float e = __expf((float)Ps[tid * 136 + j] - mx);
      sum += e; Ps[tid * 136 + j] = (half_t)e;
    }
    float inv = 1.f / sum;
    for (int j = 0; j < T; ++j) Ps[tid * 136 + j] = (half_t)((float)Ps[tid * 136 + j] * inv);
  }
  __syncthreads();
  for (int nt = 0; nt < 2; ++nt) {
    v8f acc = {};
#pragma unroll
    for (int kc = 0; kc < 4; ++kc)
      acc = wmma16(load_a_frag(Ps, 136, w * 16, kc * 32),
                   load_b_frag_kn(Vs, 40, kc * 32, nt * 16), acc);
#pragma unroll
    for (int r = 0; r < 8; ++r) {
      int row = w * 16 + r + hi * 8;
      O[((size_t)b * T + row) * 128 + h * 32 + nt * 16 + lo] = (half_t)acc[r];
    }
  }
}

// ---------------------------------------------------------------------------
// BiLSTM layer 1 (input=1, H=64). Whh staged transposed [64][260] in LDS.
// ---------------------------------------------------------------------------
__global__ void k_lstm1(const half_t* __restrict__ X, const float* __restrict__ Whh,
                        const float* __restrict__ Wih, const float* __restrict__ bih,
                        const float* __restrict__ bhh, half_t* __restrict__ H1,
                        int T, int dir) {
  extern __shared__ char smem[];
  half_t* WhhS = (half_t*)smem;        // [64][260]  ([k][n])
  half_t* hBuf = WhhS + 64 * 260;      // [16][64]
  float*  xcol = (float*)(hBuf + 16 * 64);
  int tid = threadIdx.x, mb = blockIdx.x * 16;
  for (int e = tid; e < 256 * 64; e += 128) {
    int k = e >> 8, n = e & 255;
    WhhS[k * 260 + n] = (half_t)Whh[n * 64 + k];
  }
  for (int e = tid; e < 16 * 64; e += 128) hBuf[e] = (half_t)0.f;
  int w = tid >> 5, lane = tid & 31, hi = lane >> 4, lo = lane & 15;
  float wih[4], bs[4];
#pragma unroll
  for (int g = 0; g < 4; ++g) {
    int j = g * 64 + w * 16 + lo;
    wih[g] = Wih[j]; bs[g] = bih[j] + bhh[j];
  }
  float cst[8];
#pragma unroll
  for (int r = 0; r < 8; ++r) cst[r] = 0.f;
  __syncthreads();

  for (int tt = 0; tt < T; ++tt) {
    int t = dir ? (T - 1 - tt) : tt;
    if (tid < 16) xcol[tid] = (float)X[(size_t)(mb + tid) * T + t];
    v16h a0 = load_a_frag(hBuf, 64, 0, 0);
    v16h a1 = load_a_frag(hBuf, 64, 0, 32);
    v8f gv[4];
#pragma unroll
    for (int g = 0; g < 4; ++g) {
      int n0 = g * 64 + w * 16;
      v8f acc = {};
      acc = wmma16(a0, load_b_frag_kn(WhhS, 260, 0, n0), acc);
      acc = wmma16(a1, load_b_frag_kn(WhhS, 260, 32, n0), acc);
      gv[g] = acc;
    }
    __syncthreads();
#pragma unroll
    for (int r = 0; r < 8; ++r) {
      int row = r + hi * 8;
      float xv = xcol[row];
      float iv = sigf(gv[0][r] + xv * wih[0] + bs[0]);
      float fv = sigf(gv[1][r] + xv * wih[1] + bs[1]);
      float gg = tanhf(gv[2][r] + xv * wih[2] + bs[2]);
      float ov = sigf(gv[3][r] + xv * wih[3] + bs[3]);
      float c = fv * cst[r] + iv * gg; cst[r] = c;
      float hv = ov * tanhf(c);
      hBuf[row * 64 + w * 16 + lo] = (half_t)hv;
      H1[((size_t)(mb + row) * T + t) * 128 + dir * 64 + w * 16 + lo] = (half_t)hv;
    }
    __syncthreads();
  }
}

// ---------------------------------------------------------------------------
// BiLSTM layer 2 (input=128, H=64). x tile [16][128] prefetched each step via
// TDM tensor_load_to_lds into a double buffer (wave 0 issues, all waves sync
// on s_wait_tensorcnt + barrier). Weights staged transposed [K][N].
// LDS carve (bytes): WihS 0, WhhS 66560, hBuf 99840, xB0 101888, xB1 105984.
// ---------------------------------------------------------------------------
#define L2_XB0 101888u
#define L2_XB1 105984u
__global__ void k_lstm2(const half_t* __restrict__ H1, const float* __restrict__ Whh,
                        const float* __restrict__ Wih, const float* __restrict__ bih,
                        const float* __restrict__ bhh, half_t* __restrict__ H2,
                        int T, int dir) {
  extern __shared__ char smem[];
  half_t* WihS = (half_t*)smem;          // [128][260] ([k][n])
  half_t* WhhS = WihS + 128 * 260;       // [64][260]  ([k][n])
  half_t* hBuf = WhhS + 64 * 260;        // [16][64]
  half_t* xB0  = hBuf + 16 * 64;         // [16][128] tile buffer 0
  half_t* xB1  = xB0 + 16 * 128;         // [16][128] tile buffer 1
  int tid = threadIdx.x, mb = blockIdx.x * 16;
  for (int e = tid; e < 256 * 128; e += 128) {
    int k = e >> 8, n = e & 255;
    WihS[k * 260 + n] = (half_t)Wih[n * 128 + k];
  }
  for (int e = tid; e < 256 * 64; e += 128) {
    int k = e >> 8, n = e & 255;
    WhhS[k * 260 + n] = (half_t)Whh[n * 64 + k];
  }
  for (int e = tid; e < 16 * 64; e += 128) hBuf[e] = (half_t)0.f;
  int w = tid >> 5, lane = tid & 31, hi = lane >> 4, lo = lane & 15;
  float bs[4];
#pragma unroll
  for (int g = 0; g < 4; ++g) { int j = g * 64 + w * 16 + lo; bs[g] = bih[j] + bhh[j]; }
  float cst[8];
#pragma unroll
  for (int r = 0; r < 8; ++r) cst[r] = 0.f;

  // prologue: TDM-load first x tile into buffer 0
  int t0 = dir ? (T - 1) : 0;
  if (tid < 32) {
    tdm_load_16x128(H1 + ((size_t)mb * T + t0) * 128, L2_XB0, (unsigned)T * 128);
    __builtin_amdgcn_s_wait_tensorcnt(0);
  }
  __syncthreads();

  int cur = 0;
  for (int tt = 0; tt < T; ++tt) {
    int t = dir ? (T - 1 - tt) : tt;
    // prefetch next tile into the other buffer while computing this step
    if (tt + 1 < T && tid < 32) {
      int tn = dir ? (T - 2 - tt) : (tt + 1);
      tdm_load_16x128(H1 + ((size_t)mb * T + tn) * 128,
                      cur ? L2_XB0 : L2_XB1, (unsigned)T * 128);
    }
    half_t* xBuf = cur ? xB1 : xB0;
    v16h xa0 = load_a_frag(xBuf, 128, 0, 0),  xa1 = load_a_frag(xBuf, 128, 0, 32);
    v16h xa2 = load_a_frag(xBuf, 128, 0, 64), xa3 = load_a_frag(xBuf, 128, 0, 96);
    v16h ha0 = load_a_frag(hBuf, 64, 0, 0),   ha1 = load_a_frag(hBuf, 64, 0, 32);
    v8f gv[4];
#pragma unroll
    for (int g = 0; g < 4; ++g) {
      int n0 = g * 64 + w * 16;
      v8f acc = {};
      acc = wmma16(xa0, load_b_frag_kn(WihS, 260, 0, n0), acc);
      acc = wmma16(xa1, load_b_frag_kn(WihS, 260, 32, n0), acc);
      acc = wmma16(xa2, load_b_frag_kn(WihS, 260, 64, n0), acc);
      acc = wmma16(xa3, load_b_frag_kn(WihS, 260, 96, n0), acc);
      acc = wmma16(ha0, load_b_frag_kn(WhhS, 260, 0, n0), acc);
      acc = wmma16(ha1, load_b_frag_kn(WhhS, 260, 32, n0), acc);
      gv[g] = acc;
    }
    __syncthreads();
#pragma unroll
    for (int r = 0; r < 8; ++r) {
      int row = r + hi * 8;
      int bidx = mb + row;
      float iv = sigf(gv[0][r] + bs[0]);
      float fv = sigf(gv[1][r] + bs[1]);
      float gg = tanhf(gv[2][r] + bs[2]);
      float ov = sigf(gv[3][r] + bs[3]);
      float c = fv * cst[r] + iv * gg; cst[r] = c;
      float hv = ov * tanhf(c);
      hBuf[row * 64 + w * 16 + lo] = (half_t)hv;
      float sc = (bidx >= 2 * BATCH) ? 0.1f : 1.0f;   // /P for pos,neg streams
      H2[((size_t)bidx * T + t) * 128 + dir * 64 + w * 16 + lo] = (half_t)(hv * sc);
    }
    if (tid < 32) __builtin_amdgcn_s_wait_tensorcnt(0);
    __syncthreads();
    cur ^= 1;
  }
}

// ---------------------------------------------------------------------------
// MI-LSTM persistent kernel. WH pre-transposed to [K][N] f16; per step
// 10x (h@Wh^T) + c@wA via WMMA, gate fusion, reference softmax quirk.
// ---------------------------------------------------------------------------
__global__ void k_milstm(const half_t* __restrict__ XPRE, const half_t* __restrict__ WH,
                         const half_t* __restrict__ WA,
                         const float* __restrict__ bAY, const float* __restrict__ bAI,
                         const float* __restrict__ bAP, const float* __restrict__ bAN,
                         half_t* __restrict__ HM, int T) {
  __shared__ half_t hBuf[16 * 128];
  __shared__ half_t cBuf[16 * 128];
  __shared__ float  bAs[4 * 128];
  __shared__ float  sred[64];
  __shared__ float  Zred[16];
  __shared__ float  aw[64];
  int tid = threadIdx.x, b0 = blockIdx.x * 16;
  for (int e = tid; e < 128; e += 256) {
    bAs[e] = bAY[e]; bAs[128 + e] = bAI[e]; bAs[256 + e] = bAP[e]; bAs[384 + e] = bAN[e];
  }
  for (int e = tid; e < 16 * 128; e += 256) { hBuf[e] = (half_t)0.f; cBuf[e] = (half_t)0.f; }
  int w = tid >> 5, lane = tid & 31, hi = lane >> 4, lo = lane & 15;
  int n0 = w * 16;
  float cst[8];
#pragma unroll
  for (int r = 0; r < 8; ++r) cst[r] = 0.f;
  __syncthreads();

  for (int t = 0; t < T; ++t) {
    if (tid < 64) sred[tid] = 0.f;
    if (tid < 16) Zred[tid] = 0.f;
    __syncthreads();
    v16h ha[4], ca[4];
#pragma unroll
    for (int kc = 0; kc < 4; ++kc) {
      ha[kc] = load_a_frag(hBuf, 128, 0, kc * 32);
      ca[kc] = load_a_frag(cBuf, 128, 0, kc * 32);
    }
    // gate order: 0=F 1=O 2=CY 3=IY 4=CI 5=II 6=CP 7=IP 8=CN 9=IN
    float G[10][8];
#pragma unroll
    for (int q = 0; q < 10; ++q) {
      const half_t* Wq = WH + (size_t)q * 128 * 128;   // [k][n]
      v8f acc = {};
#pragma unroll
      for (int kc = 0; kc < 4; ++kc)
        acc = wmma16(ha[kc], load_b_frag_kn(Wq, 128, kc * 32, n0), acc);
#pragma unroll
      for (int r = 0; r < 8; ++r) {
        int row = r + hi * 8, col = n0 + lo;
        G[q][r] = acc[r] +
            (float)XPRE[((size_t)(b0 + row) * T + t) * 1280 + q * 128 + col];
      }
    }
    v8f cw = {};
#pragma unroll
    for (int kc = 0; kc < 4; ++kc)
      cw = wmma16(ca[kc], load_b_frag_kn(WA, 128, kc * 32, n0), cw);

    float sf[8], so[8], lY[8], lI[8], lP[8], lN[8];
#pragma unroll
    for (int r = 0; r < 8; ++r) {
      sf[r] = sigf(G[0][r]); so[r] = sigf(G[1][r]);
      lY[r] = tanhf(G[2][r]) * sigf(G[3][r]);
      lI[r] = tanhf(G[4][r]) * sigf(G[5][r]);
      lP[r] = tanhf(G[6][r]) * sigf(G[7][r]);
      lN[r] = tanhf(G[8][r]) * sigf(G[9][r]);
      int row = r + hi * 8;
      atomicAdd(&sred[row],      lY[r] * cw[r]);
      atomicAdd(&sred[16 + row], lI[r] * cw[r]);
      atomicAdd(&sred[32 + row], lP[r] * cw[r]);
      atomicAdd(&sred[48 + row], lN[r] * cw[r]);
    }
    __syncthreads();
    {   // softmax denominator over the full [B, 4*128] (reference quirk)
      int row = tid & 15, j0 = tid >> 4;
      float part = 0.f;
      for (int m = 0; m < 4; ++m) {
        float s = sred[m * 16 + row];
        for (int j = j0; j < 128; j += 16) part += __expf(tanhf(s + bAs[m * 128 + j]));
      }
      atomicAdd(&Zred[row], part);
    }
    __syncthreads();
    if (tid < 64) {   // a_j = exp(tanh(sY + bAY[j]))/Z, j=0..3
      int row = tid & 15, j = tid >> 4;
      aw[row * 4 + j] = __expf(tanhf(sred[row] + bAs[j])) / Zred[row];
    }
    __syncthreads();
#pragma unroll
    for (int r = 0; r < 8; ++r) {
      int row = r + hi * 8, col = n0 + lo;
      float lT = aw[row * 4] * lY[r] + aw[row * 4 + 1] * lI[r] +
                 aw[row * 4 + 2] * lP[r] + aw[row * 4 + 3] * lN[r];
      float c = cst[r] * sf[r] + lT; cst[r] = c;
      float hv = tanhf(c) * so[r];
      hBuf[row * 128 + col] = (half_t)hv;
      cBuf[row * 128 + col] = (half_t)c;
      HM[((size_t)(b0 + row) * T + t) * 128 + col] = (half_t)hv;
    }
    __syncthreads();
  }
}

// ---------------------------------------------------------------------------
// small elementwise / reduction kernels
// ---------------------------------------------------------------------------
__global__ void k_pack(const float* __restrict__ y, const float* __restrict__ ix,
                       const float* __restrict__ pos, const float* __restrict__ neg,
                       half_t* __restrict__ X, int P) {
  size_t id = (size_t)blockIdx.x * 256 + threadIdx.x;
  size_t tot = (size_t)SB * SEQ;
  if (id >= tot) return;
  size_t per = (size_t)BATCH * SEQ;
  int s = (int)(id / per);
  size_t r = id % per;
  float v;
  if (s == 0) v = y[r]; else if (s == 1) v = ix[r];
  else if (s == 2) v = pos[r * P]; else v = neg[r * P];
  X[id] = (half_t)v;
}

__global__ void k_cvt(const float* __restrict__ src, int ld, int off,
                      half_t* __restrict__ dst, int n, int cols) {
  int id = blockIdx.x * 256 + threadIdx.x;
  if (id >= n) return;
  int r = id / cols, c = id % cols;
  dst[id] = (half_t)src[(size_t)r * ld + off + c];
}

// fp32 [N][ld] -> fp16 [K][N] (transposed, for kn B-fragments)
__global__ void k_cvtT(const float* __restrict__ src, int ld, int off,
                       half_t* __restrict__ dst, int N, int K) {
  int id = blockIdx.x * 256 + threadIdx.x;
  if (id >= N * K) return;
  int n = id / K, k = id % K;
  dst[(size_t)k * N + n] = (half_t)src[(size_t)n * ld + off + k];
}

__global__ void k_res_ln(const half_t* __restrict__ A, const half_t* __restrict__ B,
                         const float* __restrict__ g, const float* __restrict__ be,
                         half_t* __restrict__ O) {
  int row = blockIdx.x, tid = threadIdx.x;   // blockDim = 128
  size_t base = (size_t)row * 128 + tid;
  float v = (float)A[base] + (float)B[base];
  float s = v, s2 = v * v;
#pragma unroll
  for (int o = 16; o > 0; o >>= 1) { s += __shfl_xor(s, o, 32); s2 += __shfl_xor(s2, o, 32); }
  __shared__ float ps[4], ps2[4];
  int w = tid >> 5, lane = tid & 31;
  if (!lane) { ps[w] = s; ps2[w] = s2; }
  __syncthreads();
  float ts = ps[0] + ps[1] + ps[2] + ps[3];
  float ts2 = ps2[0] + ps2[1] + ps2[2] + ps2[3];
  float mean = ts * (1.f / 128.f);
  float var = ts2 * (1.f / 128.f) - mean * mean;
  O[base] = (half_t)((v - mean) * rsqrtf(var + 1e-5f) * g[tid] + be[tid]);
}

__global__ void k_add(const half_t* __restrict__ a, const half_t* __restrict__ b,
                      half_t* __restrict__ o, size_t n) {
  size_t id = (size_t)blockIdx.x * 256 + threadIdx.x;
  if (id < n) o[id] = (half_t)((float)a[id] + (float)b[id]);
}

__global__ void k_bn(const half_t* __restrict__ x, const float* __restrict__ g,
                     const float* __restrict__ b, const float* __restrict__ rm,
                     const float* __restrict__ rv, half_t* __restrict__ o,
                     int n, int cols) {
  int id = blockIdx.x * 256 + threadIdx.x;
  if (id >= n) return;
  int c = id % cols;
  o[id] = (half_t)(((float)x[id] - rm[c]) * rsqrtf(rv[c] + 1e-5f) * g[c] + b[c]);
}

__global__ void k_rowdot(const half_t* __restrict__ A, int n, const float* __restrict__ v,
                         const float* __restrict__ bias, float* __restrict__ out, int rows) {
  int w = threadIdx.x >> 5, lane = threadIdx.x & 31;
  int row = blockIdx.x * 8 + w;
  if (row >= rows) return;
  float s = 0.f;
  for (int e = lane; e < n; e += 32) s += (float)A[(size_t)row * n + e] * v[e];
#pragma unroll
  for (int o = 16; o > 0; o >>= 1) s += __shfl_xor(s, o, 32);
  if (!lane) out[row] = s + (bias ? bias[0] : 0.f);
}

__global__ void k_softmaxT(const float* __restrict__ sc, float* __restrict__ beta, int T) {
  int b = blockIdx.x, tid = threadIdx.x;   // blockDim = 128
  float x = sc[(size_t)b * T + tid];
  __shared__ float red[4];
  int w = tid >> 5, lane = tid & 31;
  float m = x;
#pragma unroll
  for (int o = 16; o > 0; o >>= 1) m = fmaxf(m, __shfl_xor(m, o, 32));
  if (!lane) red[w] = m;
  __syncthreads();
  m = fmaxf(fmaxf(red[0], red[1]), fmaxf(red[2], red[3]));
  __syncthreads();
  float e = __expf(x - m), s = e;
#pragma unroll
  for (int o = 16; o > 0; o >>= 1) s += __shfl_xor(s, o, 32);
  if (!lane) red[w] = s;
  __syncthreads();
  s = red[0] + red[1] + red[2] + red[3];
  beta[(size_t)b * T + tid] = e / s;
}

__global__ void k_pool(const half_t* __restrict__ HT, const float* __restrict__ beta,
                       half_t* __restrict__ out, int T) {
  int b = blockIdx.x, e = threadIdx.x;   // blockDim = 128
  float acc = 0.f;
  for (int t = 0; t < T; ++t)
    acc += (float)HT[((size_t)b * T + t) * 128 + e] * beta[(size_t)b * T + t];
  out[(size_t)b * 128 + e] = (half_t)acc;
}

// ---------------------------------------------------------------------------
// host orchestration
// ---------------------------------------------------------------------------
static inline void gemm(hipStream_t s, const half_t* A, int K, const float* W, int ldw,
                        int woff, const float* bias, half_t* C, int ldc, int coff,
                        int M, int N, int act) {
  dim3 g(M / 32, (N + 63) / 64);
  k_gemm<<<g, 256, 0, s>>>(A, K, W, ldw, woff, bias, C, ldc, coff, M, N, act);
}

struct TrP { const float *Wi, *bi, *Wo, *bo, *W1, *b1, *W2, *b2, *g0, *be0, *g1, *be1; };

static void run_transformer(hipStream_t s, const half_t* Xin, half_t* Xout,
                            size_t R, int T, const TrP& p,
                            half_t* qkv, half_t* tA, half_t* tB, half_t* tC) {
  size_t shAttn = (2 * 128 * 40 + 32 * 136 + 128 * 136) * sizeof(half_t);
  gemm(s, Xin, 128, p.Wi, 128, 0, p.bi, qkv, 384, 0, (int)R, 384, ACT_NONE);
  k_attn<<<dim3((unsigned)(R / T) * 4), 256, shAttn, s>>>(qkv, tA, T);
  gemm(s, tA, 128, p.Wo, 128, 0, p.bo, tB, 128, 0, (int)R, 128, ACT_NONE);   // h1
  k_res_ln<<<dim3((unsigned)R), 128, 0, s>>>(tB, Xin, p.g0, p.be0, tA);      // h2
  gemm(s, tA, 128, p.W1, 128, 0, p.b1, tC, 256, 0, (int)R, 256, ACT_MISH);
  gemm(s, tC, 256, p.W2, 256, 0, p.b2, tB, 128, 0, (int)R, 128, ACT_NONE);   // h3
  k_res_ln<<<dim3((unsigned)R), 128, 0, s>>>(tB, tA, p.g1, p.be1, tB);       // h4
  k_add<<<dim3((unsigned)((R * 128 + 255) / 256)), 256, 0, s>>>(tB, Xin, Xout, R * 128);
}

extern "C" void kernel_launch(void* const* d_in, const int* in_sizes, int n_in,
                              void* d_out, int out_size, void* d_ws, size_t ws_size,
                              hipStream_t stream) {
  (void)in_sizes; (void)n_in; (void)out_size; (void)ws_size;
  auto F = [&](int i) { return (const float*)d_in[i]; };
  const int T = SEQ;

  // ---- workspace carve ----
  char* ws = (char*)d_ws; size_t off = 0;
  auto carve = [&](size_t bytes) { void* p = ws + off; off = (off + bytes + 255) & ~(size_t)255; return p; };
  half_t* X    = (half_t*)carve((size_t)SB * T * 2);
  half_t* H1   = (half_t*)carve(R1ROWS * 128 * 2);
  half_t* H2   = (half_t*)carve(R1ROWS * 128 * 2);
  half_t* FEAT = (half_t*)carve(R1ROWS * 128 * 2);
  half_t* QKV  = (half_t*)carve(R1ROWS * 384 * 2);
  half_t* TA   = (half_t*)carve(R1ROWS * 128 * 2);
  half_t* TB   = (half_t*)carve(R1ROWS * 128 * 2);
  half_t* TC   = (half_t*)carve(R1ROWS * 256 * 2);
  half_t* XPRE = (half_t*)carve(R2ROWS * 1280 * 2);
  half_t* WH16 = (half_t*)carve((size_t)10 * 128 * 128 * 2);
  half_t* WA16 = (half_t*)carve((size_t)128 * 128 * 2);
  half_t* HM   = (half_t*)carve(R2ROWS * 128 * 2);
  half_t* HT   = (half_t*)carve(R2ROWS * 128 * 2);
  float*  SC   = (float*)carve(R2ROWS * 4);
  float*  BETA = (float*)carve(R2ROWS * 4);
  half_t* PL   = (half_t*)carve((size_t)BATCH * 128 * 2);
  half_t* RG0  = (half_t*)carve((size_t)BATCH * 512 * 2);
  half_t* RG1  = (half_t*)carve((size_t)BATCH * 512 * 2);

  // ---- 1. pack inputs (pos/neg use column 0) ----
  k_pack<<<dim3((SB * T + 255) / 256), 256, 0, stream>>>(F(0), F(1), F(2), F(3), X, 10);

  // ---- 2. BiLSTM layer 1 ----
  size_t sh1 = (64 * 260 + 16 * 64) * sizeof(half_t) + 16 * sizeof(float);
  k_lstm1<<<dim3(SB / 16), 128, sh1, stream>>>(X, F(35), F(36), F(38), F(37), H1, T, 0); // fwd
  k_lstm1<<<dim3(SB / 16), 128, sh1, stream>>>(X, F(31), F(32), F(34), F(33), H1, T, 1); // bwd

  // ---- 3. BiLSTM layer 2 (TDM x-tile prefetch; /10-scaled pos/neg out) ----
  size_t sh2 = (128 * 260 + 64 * 260 + 16 * 64 + 2 * 16 * 128) * sizeof(half_t);
  k_lstm2<<<dim3(SB / 16), 128, sh2, stream>>>(H1, F(43), F(44), F(46), F(45), H2, T, 0);
  k_lstm2<<<dim3(SB / 16), 128, sh2, stream>>>(H1, F(39), F(40), F(42), F(41), H2, T, 1);

  // ---- 4. encoder transformer (all 4 streams batched) ----
  TrP enc = { F(21), F(27), F(22), F(28), F(19), F(23), F(20), F(24), F(29), F(25), F(30), F(26) };
  run_transformer(stream, H2, FEAT, R1ROWS, T, enc, QKV, TA, TB, TC);

  // ---- 5. MI-LSTM weight prep + x-side projections ----
  // gate order: F O CY IY CI II CP IP CN IN
  const int WQ[10] = {51, 56, 50, 55, 47, 52, 49, 54, 48, 53};
  const int BQ[10] = {65, 70, 64, 69, 61, 66, 63, 68, 62, 67};
  const int SQ[10] = {0, 0, 0, 0, 1, 0, 2, 0, 3, 0};   // feature stream per gate
  for (int q = 0; q < 10; ++q)
    k_cvtT<<<dim3((128 * 128 + 255) / 256), 256, 0, stream>>>(F(WQ[q]), 256, 0,
        WH16 + (size_t)q * 128 * 128, 128, 128);                // Wh = W[:, :128] -> [k][n]
  k_cvt<<<dim3((128 * 128 + 255) / 256), 256, 0, stream>>>(F(71), 128, 0, WA16, 128 * 128, 128);
  for (int q = 0; q < 10; ++q) {
    const half_t* Aq = FEAT + (size_t)SQ[q] * R2ROWS * 128;
    gemm(stream, Aq, 128, F(WQ[q]), 256, 128, F(BQ[q]), XPRE, 1280, q * 128,
         (int)R2ROWS, 128, ACT_NONE);                           // Wx = W[:, 128:]
  }

  // ---- 6. MI-LSTM recurrence ----
  k_milstm<<<dim3(BATCH / 16), 256, 0, stream>>>(XPRE, WH16, WA16,
      F(60), F(57), F(59), F(58), HM, T);

  // ---- 7. attention pooling ----
  TrP ptr = { F(8), F(14), F(9), F(15), F(6), F(10), F(7), F(11), F(16), F(12), F(17), F(13) };
  run_transformer(stream, HM, FEAT, R2ROWS, T, ptr, QKV, TA, TB, TC);  // reuse FEAT
  gemm(stream, FEAT, 128, F(4), 128, 0, F(5), HT, 128, 0, (int)R2ROWS, 128, ACT_TANH);
  k_rowdot<<<dim3((unsigned)((R2ROWS + 7) / 8)), 256, 0, stream>>>(HT, 128, F(18), nullptr, SC, (int)R2ROWS);
  k_softmaxT<<<dim3(BATCH), 128, 0, stream>>>(SC, BETA, T);
  k_pool<<<dim3(BATCH), 128, 0, stream>>>(HT, BETA, PL, T);

  // ---- 8. regressor ----
  k_bn<<<dim3((BATCH * 128 + 255) / 256), 256, 0, stream>>>(PL, F(91), F(90), F(92), F(93), RG0, BATCH * 128, 128);
  gemm(stream, RG0, 128, F(76), 128, 0, F(77), RG1, 512, 0, BATCH, 512, ACT_MISH);
  k_bn<<<dim3((BATCH * 512 + 255) / 256), 256, 0, stream>>>(RG1, F(73), F(72), F(74), F(75), RG1, BATCH * 512, 512);
  gemm(stream, RG1, 512, F(82), 512, 0, F(83), RG0, 256, 0, BATCH, 256, ACT_MISH);
  k_bn<<<dim3((BATCH * 256 + 255) / 256), 256, 0, stream>>>(RG0, F(79), F(78), F(80), F(81), RG0, BATCH * 256, 256);
  gemm(stream, RG0, 256, F(88), 256, 0, F(89), RG1, 128, 0, BATCH, 128, ACT_MISH);
  k_bn<<<dim3((BATCH * 128 + 255) / 256), 256, 0, stream>>>(RG1, F(85), F(84), F(86), F(87), RG1, BATCH * 128, 128);
  k_rowdot<<<dim3((BATCH + 7) / 8), 256, 0, stream>>>(RG1, 128, F(94), F(95), (float*)d_out, BATCH);
}
